// GatheringLoss_541165879319
// MI455X (gfx1250) — compile-verified, matrix operand order checked by reading, not run on hardware
//
#include <hip/hip_runtime.h>

typedef __attribute__((ext_vector_type(16))) __bf16 v16bf;
typedef __attribute__((ext_vector_type(8)))  float  v8f;

#define C_DIM   512
#define M_ITEMS 2048
#define NQ      65536          // 64 * 1024 query rows
#define NT_TILES 128           // 2048 / 16 item tiles
#define KSTEPS   16            // 512 / 32 k-steps
#define TILE_ELEMS (16 * 32 * 16)   // bf16 elements per packed item-tile chunk (8192)

// ---------------------------------------------------------------------------
// Kernel 1: convert items f32 -> bf16, pre-swizzled into the per-lane WMMA
// B-matrix element order (ISA 7.12.2, 16-bit matrix layout), so the main
// kernel reads B as two coalesced b128 loads per lane per k-step.
// Packed index: (((nt*16 + kt)*32 + lane)*16 + e)
//   n = nt*16 + (lane&15);  K = e + (e>=8 ? 8 : 0) + (lane>>4)*8;  k = kt*32+K
// ---------------------------------------------------------------------------
__global__ void pack_items_kernel(const float* __restrict__ items,
                                  __bf16* __restrict__ bp) {
    int t    = blockIdx.x * blockDim.x + threadIdx.x;   // 0 .. 1048575
    int e    = t & 15;
    int lane = (t >> 4) & 31;
    int kt   = (t >> 9) & 15;
    int nt   = t >> 13;
    int hf   = lane >> 4;
    int K    = e + ((e >= 8) ? 8 : 0) + hf * 8;
    int k    = kt * 32 + K;
    int n    = nt * 16 + (lane & 15);
    bp[t] = (__bf16)items[n * C_DIM + k];
}

// ---------------------------------------------------------------------------
// Kernel 2: main WMMA argmax + SSE kernel.
// Block = 256 threads = 8 waves; each wave owns 16 query rows.
// Grid = 512 blocks covers 65536 rows.
// Two independent WMMA accumulator chains (even/odd k) expose ILP so the
// scheduler can overlap ds_load latency with the other chain's WMMAs.
// ---------------------------------------------------------------------------
__launch_bounds__(256, 1)
__global__ void gather_loss_main(const float* __restrict__ queries,
                                 const float* __restrict__ items,
                                 const __bf16* __restrict__ bp,
                                 float* __restrict__ partials) {
    __shared__ __align__(32) __bf16 Bs[2][TILE_ELEMS];   // 2 x 16 KB double buffer
    __shared__ float wsum[8];

    const int tid   = threadIdx.x;
    const int lane  = tid & 31;
    const int wv    = tid >> 5;
    const int hf    = lane >> 4;     // half-wave select
    const int mn    = lane & 15;     // row (A) / column (B,D) within 16
    const int qbase = blockIdx.x * 128 + wv * 16;

    // ---- Load A tile (16 rows x 512 k) once, convert f32->bf16, keep in
    // registers in the per-lane WMMA A layout: a[kt][e], K = e+(e>=8?8:0)+hf*8
    v16bf a[KSTEPS];
    {
        const float* qr = queries + (size_t)(qbase + mn) * C_DIM;
#pragma unroll
        for (int kt = 0; kt < KSTEPS; ++kt) {
#pragma unroll
            for (int e = 0; e < 16; ++e) {
                const int K = e + ((e >= 8) ? 8 : 0) + hf * 8;
                a[kt][e] = (__bf16)qr[kt * 32 + K];
            }
        }
    }

    float best_val[8];
    int   best_idx[8];
#pragma unroll
    for (int r = 0; r < 8; ++r) { best_val[r] = -3.4e38f; best_idx[r] = 0; }

    const uint4* gsrc = (const uint4*)bp;       // 1024 uint4 per item-tile chunk
    uint4* lds0 = (uint4*)&Bs[0][0];
    uint4* lds1 = (uint4*)&Bs[1][0];

    // prologue: stage tile 0
    {
        const uint4* s = gsrc + (size_t)tid * 4;
        uint4* d = lds0 + (size_t)tid * 4;
        d[0] = s[0]; d[1] = s[1]; d[2] = s[2]; d[3] = s[3];
    }
    __syncthreads();

    for (int nt = 0; nt < NT_TILES; ++nt) {
        const int cur = nt & 1;

        // prefetch next chunk into registers (overlaps with WMMA below;
        // avoids the in-order ds_store -> ds_load stall)
        uint4 p0, p1, p2, p3;
        const bool pf = (nt + 1 < NT_TILES);
        if (pf) {
            const uint4* s = gsrc + (size_t)(nt + 1) * 1024 + (size_t)tid * 4;
            p0 = s[0]; p1 = s[1]; p2 = s[2]; p3 = s[3];
        }

        // ---- 16x16 score tile: D = A(16x512) x B(512x16) via 16 WMMAs,
        // split into two independent accumulator chains for ILP.
        v8f d0 = {0.f, 0.f, 0.f, 0.f, 0.f, 0.f, 0.f, 0.f};
        v8f d1 = {0.f, 0.f, 0.f, 0.f, 0.f, 0.f, 0.f, 0.f};
#pragma unroll
        for (int kt = 0; kt < KSTEPS; kt += 2) {
            v16bf b0 = *(const v16bf*)(&Bs[cur][((kt + 0) * 32 + lane) * 16]);
            v16bf b1 = *(const v16bf*)(&Bs[cur][((kt + 1) * 32 + lane) * 16]);
            d0 = __builtin_amdgcn_wmma_f32_16x16x32_bf16(
                     false, a[kt + 0], false, b0, (short)0, d0, false, false);
            d1 = __builtin_amdgcn_wmma_f32_16x16x32_bf16(
                     false, a[kt + 1], false, b1, (short)0, d1, false, false);
        }
        const v8f d = d0 + d1;

        // ---- running argmax: D layout => lane%16 = item column,
        // VGPR r holds row r (lanes 0-15) / row r+8 (lanes 16-31)
        const int gidx = nt * 16 + mn;
#pragma unroll
        for (int r = 0; r < 8; ++r) {
            const float v = d[r];
            if (v > best_val[r]) { best_val[r] = v; best_idx[r] = gidx; }
        }

        // commit prefetched chunk into the other buffer
        if (pf) {
            uint4* dst = (cur ? lds0 : lds1) + (size_t)tid * 4;
            dst[0] = p0; dst[1] = p1; dst[2] = p2; dst[3] = p3;
        }
        __syncthreads();
    }

    // ---- cross-lane argmax reduction within each 16-lane half
    // (lowest index wins ties, matching jnp.argmax)
#pragma unroll
    for (int m = 1; m <= 8; m <<= 1) {
#pragma unroll
        for (int r = 0; r < 8; ++r) {
            const float ov = __shfl_xor(best_val[r], m, 32);
            const int   oi = __shfl_xor(best_idx[r], m, 32);
            if (ov > best_val[r] || (ov == best_val[r] && oi < best_idx[r])) {
                best_val[r] = ov; best_idx[r] = oi;
            }
        }
    }

    // ---- exact f32 SSE against the matched item, cooperatively per wave
    float acc = 0.f;
#pragma unroll
    for (int row = 0; row < 16; ++row) {
        const int r   = row & 7;
        const int src = (row < 8) ? 0 : 16;
        const int idx = __shfl(best_idx[r], src, 32);
        const float* qr = queries + (size_t)(qbase + row) * C_DIM;
        const float* ir = items   + (size_t)idx * C_DIM;
#pragma unroll 4
        for (int j = 0; j < 16; ++j) {
            const int k = lane + j * 32;
            const float dlt = qr[k] - ir[k];
            acc = fmaf(dlt, dlt, acc);
        }
    }
#pragma unroll
    for (int m = 16; m >= 1; m >>= 1) acc += __shfl_xor(acc, m, 32);
    if (lane == 0) wsum[wv] = acc;
    __syncthreads();
    if (tid == 0) {
        float s = 0.f;
#pragma unroll
        for (int w = 0; w < 8; ++w) s += wsum[w];
        partials[blockIdx.x] = s;
    }
}

// ---------------------------------------------------------------------------
// Kernel 3: deterministic fixed-order reduction of 512 partials -> mean
// ---------------------------------------------------------------------------
__global__ void reduce_partials(const float* __restrict__ partials,
                                float* __restrict__ out) {
    __shared__ float sm[256];
    const int t = threadIdx.x;
    sm[t] = partials[t] + partials[t + 256];
    __syncthreads();
    for (int s = 128; s > 0; s >>= 1) {
        if (t < s) sm[t] += sm[t + s];
        __syncthreads();
    }
    if (t == 0) out[0] = sm[0] * (1.0f / ((float)NQ * (float)C_DIM));
}

// ---------------------------------------------------------------------------
extern "C" void kernel_launch(void* const* d_in, const int* in_sizes, int n_in,
                              void* d_out, int out_size, void* d_ws, size_t ws_size,
                              hipStream_t stream) {
    const float* queries = (const float*)d_in[0];   // [64,1024,512] f32
    const float* items   = (const float*)d_in[1];   // [2048,512]    f32

    __bf16* bp       = (__bf16*)d_ws;                                 // 2 MB packed B
    float*  partials = (float*)((char*)d_ws +
                        (size_t)M_ITEMS * C_DIM * sizeof(__bf16));    // 512 floats

    pack_items_kernel<<<(M_ITEMS * C_DIM) / 256, 256, 0, stream>>>(items, bp);
    gather_loss_main<<<NQ / 128, 256, 0, stream>>>(queries, items, bp, partials);
    reduce_partials<<<1, 256, 0, stream>>>(partials, (float*)d_out);
}
